// MainBlock_40948218200278
// MI455X (gfx1250) — matrix operand order repaired and checked by reading, not compiled, not run on hardware
//
#include <hip/hip_runtime.h>
#include <math.h>

// ---------------- problem constants ----------------
constexpr int nB = 32, nN = 577, nC = 1024, nH = 16, nD = 64;
constexpr int nKEEP = 404;           // ceil(0.7 * 576)
constexpr int nN2 = 1 + nKEEP + 1;   // 406
constexpr int M1 = nB * nN;          // 18464
constexpr int M2 = nB * nN2;         // 12992

typedef _Float16 half_t;
typedef __attribute__((ext_vector_type(16))) _Float16 v16h;
typedef __attribute__((ext_vector_type(8)))  float    v8f;

static __device__ __forceinline__ v8f wmma16(v16h a, v16h b, v8f c) {
  // emits v_wmma_f32_16x16x32_f16 (f32 accumulate)
  return __builtin_amdgcn_wmma_f32_16x16x32_f16(false, a, false, b, (short)0, c, false, false);
}

union FragU { uint4 u[2]; v16h v; };

// A-matrix 16-bit frag: row = lane&15; elems 0..7 -> K=kb..kb+7, elems 8..15 -> K=kb+16..kb+23
static __device__ __forceinline__ v16h load_afrag(const half_t* row, int kb) {
  FragU c;
  c.u[0] = *(const uint4*)(row + kb);
  c.u[1] = *(const uint4*)(row + kb + 16);
  return c.v;
}
// B-matrix 16-bit frag: col = lane&15; elems 0..7 -> K=kb..kb+7, elems 8..15 -> K=kb+16..kb+23
static __device__ __forceinline__ v16h load_bfrag(const half_t* row, int kb) {
  FragU c;
  c.u[0] = *(const uint4*)(row + kb);
  c.u[1] = *(const uint4*)(row + kb + 16);
  return c.v;
}

// ---- CDNA5 async global->LDS copy (ASYNCcnt-tracked, no VGPR data staging) ----
static __device__ __forceinline__ void async_ld_b128(void* ldsPtr, const void* gPtr) {
  unsigned lds_off = (unsigned)(unsigned long long)ldsPtr;              // low 32b = LDS byte offset
  unsigned long long ga = (unsigned long long)gPtr;
  asm volatile("global_load_async_to_lds_b128 %0, %1, off"
               :: "v"(lds_off), "v"(ga) : "memory");
}
static __device__ __forceinline__ void wait_async0() {
  asm volatile("s_wait_asynccnt 0" ::: "memory");
}

// ---------------- weight transpose fp32[K][N] -> f16[N][K] ----------------
__global__ __launch_bounds__(256) void transpose_w_kernel(
    const float* __restrict__ w, half_t* __restrict__ o, int K, int Ntot) {
  __shared__ float tile[32][33];
  int bx = blockIdx.x * 32;   // n base
  int by = blockIdx.y * 32;   // k base
  int tx = threadIdx.x & 31, ty = threadIdx.x >> 5;
  #pragma unroll
  for (int r = ty; r < 32; r += 8)
    tile[r][tx] = w[(size_t)(by + r) * Ntot + bx + tx];
  __syncthreads();
  #pragma unroll
  for (int r = ty; r < 32; r += 8)
    o[(size_t)(bx + r) * K + by + tx] = (half_t)tile[tx][r];
}

// ---------------- LayerNorm (rows of 1024) -> f16 ----------------
__global__ __launch_bounds__(256) void layernorm_f16_kernel(
    const float* __restrict__ x, const float* __restrict__ w,
    const float* __restrict__ bp, half_t* __restrict__ out) {
  int row = blockIdx.x, t = threadIdx.x;
  const float* xr = x + (size_t)row * nC;
  float4 v = *(const float4*)(xr + t * 4);
  __shared__ float red[256];
  red[t] = v.x + v.y + v.z + v.w;
  __syncthreads();
  for (int st = 128; st > 0; st >>= 1) { if (t < st) red[t] += red[t + st]; __syncthreads(); }
  float mu = red[0] * (1.0f / nC);
  __syncthreads();
  float dx = v.x - mu, dy = v.y - mu, dz = v.z - mu, dw = v.w - mu;
  red[t] = dx * dx + dy * dy + dz * dz + dw * dw;
  __syncthreads();
  for (int st = 128; st > 0; st >>= 1) { if (t < st) red[t] += red[t + st]; __syncthreads(); }
  float rstd = rsqrtf(red[0] * (1.0f / nC) + 1e-5f);
  float4 wv = *(const float4*)(w + t * 4);
  float4 bv = *(const float4*)(bp + t * 4);
  half_t* orow = out + (size_t)row * nC + t * 4;
  orow[0] = (half_t)(dx * rstd * wv.x + bv.x);
  orow[1] = (half_t)(dy * rstd * wv.y + bv.y);
  orow[2] = (half_t)(dz * rstd * wv.z + bv.z);
  orow[3] = (half_t)(dw * rstd * wv.w + bv.w);
}

// ---------------- generic WMMA GEMM: out = A[MxK] * W[NxK]^T, epilogue by MODE --------
// Double-buffered LDS, tiles filled by async global->LDS copies overlapped with WMMA.
// MODE 0: scatter to q/k/v [B,H,N,D] f16 (q pre-scaled by 1/8)
// MODE 1: outF = resid + acc + bias (f32)
// MODE 2: outH = gelu(acc + bias) (f16, exact erf gelu)
// MODE 3: outF += acc + bias (f32, in-place residual in d_out)
template<int MODE>
__global__ __launch_bounds__(128) void gemm_f16_kernel(
    const half_t* __restrict__ A, const half_t* __restrict__ W,
    int M, int Ntot, int K,
    const float* __restrict__ bias, const float* __restrict__ resid,
    float* __restrict__ outF, half_t* __restrict__ outH,
    half_t* __restrict__ qo, half_t* __restrict__ ko, half_t* __restrict__ vo) {
  __shared__ half_t sA[2][64][40];
  __shared__ half_t sB[2][64][40];
  int t = threadIdx.x;
  int wave = t >> 5, lane = t & 31;
  int wm = wave >> 1, wn = wave & 1;       // 2x2 waves -> 64x64 block tile
  int lhalf = lane >> 4, l16 = lane & 15;
  int tileM = blockIdx.x * 64, tileN = blockIdx.y * 64;
  v8f zero = {0.f,0.f,0.f,0.f,0.f,0.f,0.f,0.f};
  v8f acc[2][2];
  acc[0][0] = zero; acc[0][1] = zero; acc[1][0] = zero; acc[1][1] = zero;

  auto issue_tile = [&](int k0, int buf) {
    #pragma unroll
    for (int u0 = 0; u0 < 2; ++u0) {          // 128 thr x 2 = 256 x b128
      int u = t + u0 * 128;
      int r = u >> 2, kc = (u & 3) << 3;
      int gm = tileM + r; if (gm >= M) gm = M - 1;
      async_ld_b128(&sA[buf][r][kc], A + (size_t)gm * K + k0 + kc);
      int gn = tileN + r;
      async_ld_b128(&sB[buf][r][kc], W + (size_t)gn * K + k0 + kc);
    }
  };

  const int nIter = K >> 5;
  issue_tile(0, 0);
  wait_async0();
  __syncthreads();

  for (int it = 0; it < nIter; ++it) {
    int buf = it & 1;
    if (it + 1 < nIter) issue_tile((it + 1) << 5, buf ^ 1);  // overlap copy with compute
    v16h af[2], bfr[2];
    #pragma unroll
    for (int mt = 0; mt < 2; ++mt) af[mt] = load_afrag(&sA[buf][wm * 32 + mt * 16 + l16][0], lhalf * 8);
    #pragma unroll
    for (int nt = 0; nt < 2; ++nt) bfr[nt] = load_bfrag(&sB[buf][wn * 32 + nt * 16 + l16][0], lhalf * 8);
    #pragma unroll
    for (int mt = 0; mt < 2; ++mt)
      #pragma unroll
      for (int nt = 0; nt < 2; ++nt)
        acc[mt][nt] = wmma16(af[mt], bfr[nt], acc[mt][nt]);
    wait_async0();
    __syncthreads();
  }

  #pragma unroll
  for (int mt = 0; mt < 2; ++mt) {
    #pragma unroll
    for (int nt = 0; nt < 2; ++nt) {
      int n = tileN + wn * 32 + nt * 16 + l16;
      #pragma unroll
      for (int i = 0; i < 8; ++i) {
        int m = tileM + wm * 32 + mt * 16 + i + 8 * lhalf;
        if (m >= M) continue;
        float val = acc[mt][nt][i];
        if (MODE == 0) {
          int which = n >> 10, hd = n & 1023, h = hd >> 6, d = hd & 63;
          int b = m / nN, tok = m - b * nN;
          size_t off = ((size_t)(b * nH + h) * nN + tok) * nD + d;
          if (which == 0)      qo[off] = (half_t)(val * 0.125f);   // fold softmax scale
          else if (which == 1) ko[off] = (half_t)val;
          else                 vo[off] = (half_t)val;
        } else if (MODE == 1) {
          size_t off = (size_t)m * Ntot + n;
          outF[off] = resid[off] + val + bias[n];
        } else if (MODE == 2) {
          float u2 = val + bias[n];
          outH[(size_t)m * Ntot + n] =
              (half_t)(0.5f * u2 * (1.0f + erff(u2 * 0.70710678118654752f)));
        } else {
          size_t off = (size_t)m * Ntot + n;
          outF[off] += val + bias[n];
        }
      }
    }
  }
}

// ---------------- flash attention: grid (rowtile=10, H, B), 4 waves x 16 rows ----------
__global__ __launch_bounds__(128) void flash_attn_kernel(
    const half_t* __restrict__ q, const half_t* __restrict__ k,
    const half_t* __restrict__ v, half_t* __restrict__ out) {
  int b = blockIdx.z, h = blockIdx.y, rt = blockIdx.x;
  int t = threadIdx.x, wave = t >> 5, lane = t & 31, lhalf = lane >> 4, l16 = lane & 15;
  size_t bh = (size_t)(b * nH + h);
  const half_t* qb = q + bh * nN * nD;
  const half_t* kb = k + bh * nN * nD;
  const half_t* vb = v + bh * nN * nD;
  __shared__ half_t sK[64][72];        // [key][d]   (filled by async copies)
  __shared__ half_t sV[64][72];        // transposed: [d][key]
  __shared__ half_t sP[4][16][72];     // per-wave P tile [m][key]
  int rowbase = rt * 64 + wave * 16;
  int qrow = rowbase + l16; if (qrow > nN - 1) qrow = nN - 1;

  v16h qa[2];
  {
    const half_t* qp = qb + (size_t)qrow * nD + lhalf * 8;
    FragU c0, c1;
    c0.u[0] = *(const uint4*)(qp);      c0.u[1] = *(const uint4*)(qp + 16);
    c1.u[0] = *(const uint4*)(qp + 32); c1.u[1] = *(const uint4*)(qp + 48);
    qa[0] = c0.v; qa[1] = c1.v;
  }
  v8f zero = {0.f,0.f,0.f,0.f,0.f,0.f,0.f,0.f};
  v8f o[4]; o[0] = zero; o[1] = zero; o[2] = zero; o[3] = zero;
  float rmax[8], rsum[8];
  #pragma unroll
  for (int i = 0; i < 8; ++i) { rmax[i] = -1e30f; rsum[i] = 0.0f; }

  for (int kt = 0; kt < 10; ++kt) {
    int kbase = kt * 64;
    #pragma unroll
    for (int u0 = 0; u0 < 4; ++u0) {               // K tile: async global->LDS
      int u = t + u0 * 128;
      int r = u >> 3, kc = (u & 7) << 3;
      int gk = kbase + r; if (gk > nN - 1) gk = nN - 1;
      async_ld_b128(&sK[r][kc], kb + (size_t)gk * nD + kc);
    }
    #pragma unroll
    for (int u0 = 0; u0 < 4; ++u0) {               // V tile, transposed into LDS
      int u = t + u0 * 128;
      int r = u >> 3, dc = (u & 7) << 3;
      int gk = kbase + r; if (gk > nN - 1) gk = nN - 1;
      uint4 raw = *(const uint4*)(vb + (size_t)gk * nD + dc);
      const half_t* pr = (const half_t*)&raw;
      #pragma unroll
      for (int j = 0; j < 8; ++j) sV[dc + j][r] = pr[j];
    }
    wait_async0();
    __syncthreads();

    // S = Q * K^T (16x64 per wave): batch all B-frag loads, then WMMAs back-to-back
    v16h bS[4][2];
    #pragma unroll
    for (int nt = 0; nt < 4; ++nt)
      #pragma unroll
      for (int kc = 0; kc < 2; ++kc)
        bS[nt][kc] = load_bfrag(&sK[nt * 16 + l16][0], kc * 32 + lhalf * 8);
    v8f s4[4];
    #pragma unroll
    for (int nt = 0; nt < 4; ++nt) {
      v8f a9 = wmma16(qa[0], bS[nt][0], zero);
      s4[nt] = wmma16(qa[1], bS[nt][1], a9);
    }

    // mask invalid keys + online softmax
    float nm[8];
    #pragma unroll
    for (int i = 0; i < 8; ++i) nm[i] = rmax[i];
    #pragma unroll
    for (int nt = 0; nt < 4; ++nt) {
      int key = kbase + nt * 16 + l16;
      bool valid = key < nN;
      #pragma unroll
      for (int i = 0; i < 8; ++i) {
        float sv = valid ? s4[nt][i] : -1e30f;
        s4[nt][i] = sv;
        nm[i] = fmaxf(nm[i], sv);
      }
    }
    #pragma unroll
    for (int i = 0; i < 8; ++i) {  // row-max across 16-lane group (wave32 shuffles)
      float m2 = nm[i];
      m2 = fmaxf(m2, __shfl_xor(m2, 1, 32));
      m2 = fmaxf(m2, __shfl_xor(m2, 2, 32));
      m2 = fmaxf(m2, __shfl_xor(m2, 4, 32));
      m2 = fmaxf(m2, __shfl_xor(m2, 8, 32));
      nm[i] = m2;
    }
    float corr[8], psum[8];
    #pragma unroll
    for (int i = 0; i < 8; ++i) { corr[i] = expf(rmax[i] - nm[i]); psum[i] = 0.0f; rmax[i] = nm[i]; }
    #pragma unroll
    for (int nt = 0; nt < 4; ++nt) {
      #pragma unroll
      for (int i = 0; i < 8; ++i) {
        float p = expf(s4[nt][i] - nm[i]);
        psum[i] += p;
        sP[wave][i + 8 * lhalf][nt * 16 + l16] = (half_t)p;   // C-layout -> A-storage
      }
    }
    #pragma unroll
    for (int i = 0; i < 8; ++i) {
      float s2 = psum[i];
      s2 += __shfl_xor(s2, 1, 32);
      s2 += __shfl_xor(s2, 2, 32);
      s2 += __shfl_xor(s2, 4, 32);
      s2 += __shfl_xor(s2, 8, 32);
      rsum[i] = rsum[i] * corr[i] + s2;
      #pragma unroll
      for (int dt = 0; dt < 4; ++dt) o[dt][i] *= corr[i];
    }
    __syncthreads();   // sP visible; sV still valid

    // O += P * V : batch frag loads per K-chunk, then 4 WMMAs
    #pragma unroll
    for (int kc = 0; kc < 2; ++kc) {
      v16h pf = load_afrag(&sP[wave][l16][0], kc * 32 + lhalf * 8);
      v16h vf[4];
      #pragma unroll
      for (int dt = 0; dt < 4; ++dt)
        vf[dt] = load_bfrag(&sV[dt * 16 + l16][0], kc * 32 + lhalf * 8);
      #pragma unroll
      for (int dt = 0; dt < 4; ++dt)
        o[dt] = wmma16(pf, vf[dt], o[dt]);
    }
    __syncthreads();   // before next tile's K/V loads
  }

  #pragma unroll
  for (int i = 0; i < 8; ++i) {
    int row = rowbase + i + 8 * lhalf;
    if (row < nN) {
      float inv = 1.0f / rsum[i];
      #pragma unroll
      for (int dt = 0; dt < 4; ++dt)
        out[((size_t)b * nN + row) * nC + h * nD + dt * 16 + l16] = (half_t)(o[dt][i] * inv);
    }
  }
}

// ---------------- cls-token attention row, mean over heads (deterministic) -------------
__global__ __launch_bounds__(256) void cls_attn_kernel(
    const half_t* __restrict__ q, const half_t* __restrict__ k, float* __restrict__ cls) {
  int b = blockIdx.x, t = threadIdx.x;
  __shared__ float acc[576];
  __shared__ float lbuf[640];
  __shared__ float red[256];
  __shared__ float q0[64];
  for (int j = t; j < 576; j += 256) acc[j] = 0.0f;
  for (int h = 0; h < nH; ++h) {
    const half_t* qp = q + ((size_t)(b * nH + h) * nN) * nD;   // query row 0 (pre-scaled)
    if (t < 64) q0[t] = (float)qp[t];
    __syncthreads();
    for (int j = t; j < nN; j += 256) {
      const half_t* kp = k + ((size_t)(b * nH + h) * nN + j) * nD;
      float s = 0.0f;
      #pragma unroll
      for (int d = 0; d < nD; ++d) s += q0[d] * (float)kp[d];
      lbuf[j] = s;
    }
    __syncthreads();
    float mx = -1e30f;
    for (int j = t; j < nN; j += 256) mx = fmaxf(mx, lbuf[j]);
    red[t] = mx; __syncthreads();
    for (int st = 128; st > 0; st >>= 1) { if (t < st) red[t] = fmaxf(red[t], red[t + st]); __syncthreads(); }
    float m = red[0];
    __syncthreads();
    float ls = 0.0f;
    for (int j = t; j < nN; j += 256) { float p = expf(lbuf[j] - m); lbuf[j] = p; ls += p; }
    red[t] = ls; __syncthreads();
    for (int st = 128; st > 0; st >>= 1) { if (t < st) red[t] += red[t + st]; __syncthreads(); }
    float inv = 1.0f / (red[0] * (float)nH);
    __syncthreads();
    for (int j = t; j < 576; j += 256) acc[j] += lbuf[j + 1] * inv;
    __syncthreads();
  }
  for (int j = t; j < 576; j += 256) cls[(size_t)b * 576 + j] = acc[j];
}

// ---------------- top-k via bitonic sort (val desc, idx asc = jax.lax.top_k) -----------
__global__ __launch_bounds__(512) void topk_kernel(
    const float* __restrict__ cls, int* __restrict__ idxout,
    unsigned char* __restrict__ flags, float* __restrict__ outtail) {
  int b = blockIdx.x, t = threadIdx.x;
  __shared__ float sv[1024];
  __shared__ int   si[1024];
  for (int j = t; j < 1024; j += 512) {
    if (j < 576) { sv[j] = cls[(size_t)b * 576 + j]; si[j] = j; }
    else         { sv[j] = -1e30f; si[j] = 1000000 + j; }
  }
  __syncthreads();
  for (int ks = 2; ks <= 1024; ks <<= 1) {
    for (int jj = ks >> 1; jj > 0; jj >>= 1) {
      for (int e = t; e < 1024; e += 512) {
        int p = e ^ jj;
        if (p > e) {
          bool desc = ((e & ks) == 0);
          float va = sv[e], vb = sv[p];
          int ia = si[e], ib = si[p];
          bool aGreater = (va > vb) || (va == vb && ia < ib);
          if (desc ? !aGreater : aGreater) { sv[e] = vb; sv[p] = va; si[e] = ib; si[p] = ia; }
        }
      }
      __syncthreads();
    }
  }
  for (int j = t; j < 576; j += 512) flags[b * 576 + j] = 0;
  __syncthreads();
  for (int j = t; j < nKEEP; j += 512) {
    int id = si[j];
    idxout[b * nKEEP + j] = id;
    outtail[1 + b * nKEEP + j] = (float)id;   // idx output (as float)
    flags[b * 576 + id] = 1;
  }
  if (b == 0 && t == 0) outtail[0] = (float)(nN2 - 1);   // n_tokens = 405
}

// ---------------- token gather + fused extra token -> d_out x2 -------------------------
__global__ __launch_bounds__(256) void gather_kernel(
    const float* __restrict__ xres, const float* __restrict__ cls,
    const int* __restrict__ idx, const unsigned char* __restrict__ flags,
    float* __restrict__ x2) {
  int s = blockIdx.x, b = blockIdx.y, t = threadIdx.x;
  float* orow = x2 + ((size_t)b * nN2 + s) * nC;
  if (s == 0) {
    const float* ir = xres + (size_t)b * nN * nC;
    for (int c = t; c < nC; c += 256) orow[c] = ir[c];
  } else if (s <= nKEEP) {
    int tok = idx[b * nKEEP + (s - 1)] + 1;
    const float* ir = xres + ((size_t)b * nN + tok) * nC;
    for (int c = t; c < nC; c += 256) orow[c] = ir[c];
  } else {
    for (int c = t; c < nC; c += 256) {
      float a = 0.0f;
      for (int j = 0; j < 576; ++j) {
        if (!flags[b * 576 + j])
          a += xres[((size_t)b * nN + 1 + j) * nC + c] * cls[(size_t)b * 576 + j];
      }
      orow[c] = a;
    }
  }
}

// ---------------- host-side orchestration ----------------------------------------------
extern "C" void kernel_launch(void* const* d_in, const int* in_sizes, int n_in,
                              void* d_out, int out_size, void* d_ws, size_t ws_size,
                              hipStream_t stream) {
  const float* x      = (const float*)d_in[0];
  const float* qkv_w  = (const float*)d_in[1];
  const float* proj_w = (const float*)d_in[2];
  const float* proj_b = (const float*)d_in[3];
  const float* n1w    = (const float*)d_in[4];
  const float* n1b    = (const float*)d_in[5];
  const float* n2w    = (const float*)d_in[6];
  const float* n2b    = (const float*)d_in[7];
  const float* fc1_w  = (const float*)d_in[8];
  const float* fc1_b  = (const float*)d_in[9];
  const float* fc2_w  = (const float*)d_in[10];
  const float* fc2_b  = (const float*)d_in[11];
  float* outx = (float*)d_out;

  // workspace layout (~177 MB, with reuse)
  half_t* wQkvT  = (half_t*)d_ws;                         // [3072][1024]
  half_t* wProjT = wQkvT  + (size_t)3072 * 1024;          // [1024][1024]
  half_t* wFc1T  = wProjT + (size_t)1024 * 1024;          // [4096][1024]
  half_t* wFc2T  = wFc1T  + (size_t)4096 * 1024;          // [1024][4096]
  half_t* region1 = wFc2T + (size_t)1024 * 4096;          // 18,907,136 halfs
  half_t* h1       = region1;                             // LN1 out (dead after QKV)
  half_t* attn_out = region1;                             // reuse (dead after proj)
  half_t* h2       = region1;                             // reuse (LN2 out)
  const size_t QKVSZ = (size_t)nB * nH * nN * nD;         // 18,907,136
  half_t* qbuf = region1 + (size_t)M1 * nC;
  half_t* kbuf = qbuf + QKVSZ;
  half_t* vbuf = kbuf + QKVSZ;
  float*  xres   = (float*)qbuf;    // reuse region2 after attention (75.6 MB)
  half_t* fc1out = qbuf;            // reuse region2 after gather (106.4 MB)
  float* clsbuf = (float*)(vbuf + QKVSZ);
  int* idxbuf = (int*)(clsbuf + nB * 576);
  unsigned char* flagbuf = (unsigned char*)(idxbuf + nB * nKEEP);
  float* tail = outx + (size_t)nB * nN2 * nC;             // n_tokens + idx

  // 1. weight convert+transpose to f16 [N][K]
  transpose_w_kernel<<<dim3(3072 / 32, 1024 / 32), 256, 0, stream>>>(qkv_w, wQkvT, 1024, 3072);
  transpose_w_kernel<<<dim3(1024 / 32, 1024 / 32), 256, 0, stream>>>(proj_w, wProjT, 1024, 1024);
  transpose_w_kernel<<<dim3(4096 / 32, 1024 / 32), 256, 0, stream>>>(fc1_w, wFc1T, 1024, 4096);
  transpose_w_kernel<<<dim3(1024 / 32, 4096 / 32), 256, 0, stream>>>(fc2_w, wFc2T, 4096, 1024);
  // 2. LN1
  layernorm_f16_kernel<<<M1, 256, 0, stream>>>(x, n1w, n1b, h1);
  // 3. QKV GEMM (WMMA + async double-buffered tiles) with scatter epilogue
  gemm_f16_kernel<0><<<dim3(289, 48), 128, 0, stream>>>(
      h1, wQkvT, M1, 3072, 1024, nullptr, nullptr, nullptr, nullptr, qbuf, kbuf, vbuf);
  // 4. flash attention (WMMA, async K tiles)
  flash_attn_kernel<<<dim3(10, nH, nB), 128, 0, stream>>>(qbuf, kbuf, vbuf, attn_out);
  // 5. cls attention probabilities
  cls_attn_kernel<<<nB, 256, 0, stream>>>(qbuf, kbuf, clsbuf);
  // 6. proj GEMM + bias + residual -> xres (f32)
  gemm_f16_kernel<1><<<dim3(289, 16), 128, 0, stream>>>(
      attn_out, wProjT, M1, 1024, 1024, proj_b, x, xres, nullptr, nullptr, nullptr, nullptr);
  // 7. top-k + tail outputs
  topk_kernel<<<nB, 512, 0, stream>>>(clsbuf, idxbuf, flagbuf, tail);
  // 8. gather tokens + fused extra token -> d_out
  gather_kernel<<<dim3(nN2, nB), 256, 0, stream>>>(xres, clsbuf, idxbuf, flagbuf, outx);
  // 9. LN2 on pruned tokens
  layernorm_f16_kernel<<<M2, 256, 0, stream>>>(outx, n2w, n2b, h2);
  // 10. fc1 GEMM + bias + exact GELU -> f16
  gemm_f16_kernel<2><<<dim3(203, 64), 128, 0, stream>>>(
      h2, wFc1T, M2, 4096, 1024, fc1_b, nullptr, nullptr, fc1out, nullptr, nullptr, nullptr);
  // 11. fc2 GEMM + bias, accumulated into d_out (residual)
  gemm_f16_kernel<3><<<dim3(203, 16), 128, 0, stream>>>(
      fc1out, wFc2T, M2, 1024, 4096, fc2_b, nullptr, outx, nullptr, nullptr, nullptr, nullptr);
}
